// Carafe_19344532701312
// MI455X (gfx1250) — matrix operand order repaired and checked by reading, not compiled, not run on hardware
//
#include <hip/hip_runtime.h>

// CDNA5 / gfx1250. wave32. WMMA f16 -> f32 accumulate for both 1x1-conv GEMMs.
typedef __attribute__((ext_vector_type(16))) _Float16 v16h;
typedef __attribute__((ext_vector_type(8)))  float    v8f;
typedef __attribute__((ext_vector_type(2)))  float    v2f;

#define EPSV   1e-5f
#define C_IN   256
#define OCH    64
#define P_PAD  48
#define P_REAL 36
#define HWD    4096   // 64*64
#define WD     64
#define HD     64

// ---------------- workspace layout (bytes) ----------------
// 0      gamma0[64] f32
// 256    beta0 [64] f32
// 512    gamma1[48] f32
// 768    beta1 [48] f32
// 1024   w0h  [64][256] f16   (32768 B)
// 33792  w1h  [48][64]  f16   ( 6144 B)
// 40960  wts  [32768][36] f32 (4.72 MB)

// ---------------------------------------------------------------------------
// Kernel 0: fold BN into (gamma, beta), convert weights to f16 (pad P 36->48)
// ---------------------------------------------------------------------------
__global__ void carafe_prep(const float* __restrict__ w0, const float* __restrict__ b0,
                            const float* __restrict__ g0, const float* __restrict__ bb0,
                            const float* __restrict__ m0, const float* __restrict__ v0,
                            const float* __restrict__ w1, const float* __restrict__ b1,
                            const float* __restrict__ g1, const float* __restrict__ bb1,
                            const float* __restrict__ m1, const float* __restrict__ v1,
                            float* gamma0, float* beta0, float* gamma1, float* beta1,
                            _Float16* w0h, _Float16* w1h) {
    const int t = threadIdx.x;
    if (t < OCH) {
        float s = g0[t] * rsqrtf(v0[t] + EPSV);
        gamma0[t] = s;
        beta0[t]  = b0[t] * s + bb0[t] - m0[t] * s;
    }
    if (t < P_PAD) {
        if (t < P_REAL) {
            float s = g1[t] * rsqrtf(v1[t] + EPSV);
            gamma1[t] = s;
            beta1[t]  = b1[t] * s + bb1[t] - m1[t] * s;
        } else {
            gamma1[t] = 0.f; beta1[t] = 0.f;
        }
    }
    for (int i = t; i < OCH * C_IN; i += 256) w0h[i] = (_Float16)w0[i];
    for (int i = t; i < P_PAD * OCH; i += 256) {
        int p = i / OCH, o = i - p * OCH;
        w1h[i] = (p < P_REAL) ? (_Float16)w1[p * OCH + o] : (_Float16)0.f;
    }
}

// ---------------------------------------------------------------------------
// Kernel 1: layer0 GEMM (WMMA) + BN/ReLU -> encoder GEMM (WMMA) + BN/ReLU
//           -> softmax over 9 taps -> wts[pixel][36]
// 256 threads = 8 wave32; each wave owns 16 consecutive pixels.
// ---------------------------------------------------------------------------
__global__ void __launch_bounds__(256)
carafe_enc(const float* __restrict__ x,
           const _Float16* __restrict__ w0h, const _Float16* __restrict__ w1h,
           const float* __restrict__ gamma0, const float* __restrict__ beta0,
           const float* __restrict__ gamma1, const float* __restrict__ beta1,
           float* __restrict__ wts) {
    __shared__ _Float16 yT[8][16][OCH];     // 16 KB, per-wave regions
    __shared__ float    eT[8][16][P_PAD];   // 24 KB, per-wave regions

    const int tid  = threadIdx.x;
    const int wave = tid >> 5;
    const int lid  = tid & 31;
    const int lo   = lid & 15;      // column / M-row index inside 16-halves
    const int hsel = lid >> 4;      // lane half select (ISA K-split)

    const int pbase = blockIdx.x * 128 + wave * 16;   // first pixel of this wave
    const int n_img = pbase >> 12;                    // / 4096 (tile never crosses n)
    const int hw0   = pbase & 4095;

    // ---------------- GEMM1: [16 x 256] @ [256 x 64] ----------------
    v8f acc[4];
#pragma unroll
    for (int nt = 0; nt < 4; ++nt) acc[nt] = (v8f){0.f,0.f,0.f,0.f,0.f,0.f,0.f,0.f};

    const float* xa = x + ((size_t)n_img * C_IN) * HWD + (size_t)(hw0 + lo);
    const int kbA = hsel * 16;      // A k order chosen here (contiguous): lane half holds K = half*16 + 0..15
#pragma unroll
    for (int kt = 0; kt < 8; ++kt) {
        const int ktile = kt * 32;
        v16h a;
#pragma unroll
        for (int i = 0; i < 8; ++i)
            a[i]     = (_Float16)xa[(size_t)(ktile + kbA + i) * HWD];
#pragma unroll
        for (int i = 0; i < 8; ++i)
            a[8 + i] = (_Float16)xa[(size_t)(ktile + kbA + 8 + i) * HWD];
#pragma unroll
        for (int nt = 0; nt < 4; ++nt) {
            // B uses the same k order as A: lane half holds K = half*16 + 0..15, contiguous in memory
            const _Float16* wb = w0h + (size_t)(nt * 16 + lo) * C_IN + ktile + (hsel << 4);
            v16h b = *(const v16h*)wb;
            acc[nt] = __builtin_amdgcn_wmma_f32_16x16x32_f16(
                false, a, false, b, (short)0, acc[nt], false, false);
        }
    }

    // BN + ReLU -> f16 tile in LDS. C/D layout: lane col N=lo, rows m = v + 8*hsel
#pragma unroll
    for (int nt = 0; nt < 4; ++nt) {
        const int o = nt * 16 + lo;
        const float g = gamma0[o], be = beta0[o];
#pragma unroll
        for (int v = 0; v < 8; ++v) {
            const int m = v + (hsel << 3);
            float val = acc[nt][v] * g + be;
            yT[wave][m][o] = (_Float16)(val > 0.f ? val : 0.f);
        }
    }
    __syncthreads();

    // ---------------- GEMM2: [16 x 64] @ [64 x 48] ----------------
    v8f acc2[3];
#pragma unroll
    for (int nt = 0; nt < 3; ++nt) acc2[nt] = (v8f){0.f,0.f,0.f,0.f,0.f,0.f,0.f,0.f};
#pragma unroll
    for (int kt = 0; kt < 2; ++kt) {
        const int ktile = kt * 32;
        v16h a;
#pragma unroll
        for (int i = 0; i < 8; ++i) a[i]     = yT[wave][lo][ktile + kbA + i];
#pragma unroll
        for (int i = 0; i < 8; ++i) a[8 + i] = yT[wave][lo][ktile + kbA + 8 + i];
#pragma unroll
        for (int nt = 0; nt < 3; ++nt) {
            const _Float16* wb = w1h + (size_t)(nt * 16 + lo) * OCH + ktile + (hsel << 4);
            v16h b = *(const v16h*)wb;
            acc2[nt] = __builtin_amdgcn_wmma_f32_16x16x32_f16(
                false, a, false, b, (short)0, acc2[nt], false, false);
        }
    }

#pragma unroll
    for (int nt = 0; nt < 3; ++nt) {
        const int pch = nt * 16 + lo;
        const float g = gamma1[pch], be = beta1[pch];
#pragma unroll
        for (int v = 0; v < 8; ++v) {
            const int m = v + (hsel << 3);
            float val = acc2[nt][v] * g + be;
            eT[wave][m][pch] = (val > 0.f ? val : 0.f);
        }
    }
    __syncthreads();

    // ---------------- softmax over k=9 taps, per (pixel m, up-pos u) ----------------
    // 64 tasks per wave, 2 per lane
#pragma unroll
    for (int tt = 0; tt < 2; ++tt) {
        const int task = lid + tt * 32;
        const int m = task >> 2;
        const int u = task & 3;
        float e[9];
        float mx = -1e30f;
#pragma unroll
        for (int k = 0; k < 9; ++k) { e[k] = eT[wave][m][k * 4 + u]; mx = fmaxf(mx, e[k]); }
        float s = 0.f;
#pragma unroll
        for (int k = 0; k < 9; ++k) { e[k] = __expf(e[k] - mx); s += e[k]; }
        const float inv = 1.f / s;
        float* wq = wts + (size_t)(pbase + m) * 36 + u;
#pragma unroll
        for (int k = 0; k < 9; ++k) wq[k * 4] = e[k] * inv;
    }
}

// ---------------------------------------------------------------------------
// Kernel 2: CARAFE reassembly + pixelshuffle. Block = (n,h); lanes along w so
// the 134 MB output stream stores coalesced pairs. 36 weights/thread in
// registers; 256 channels split across 4 lane groups. Output uses NT stores
// (write-once stream) so x (33.5 MB) and wts (4.7 MB) stay resident in the
// 192 MB L2 for the 9x patch reuse.
// ---------------------------------------------------------------------------
__global__ void __launch_bounds__(256)
carafe_apply(const float* __restrict__ x, const float* __restrict__ wts,
             float* __restrict__ out) {
    const int n  = blockIdx.x >> 6;
    const int h  = blockIdx.x & 63;
    const int w  = threadIdx.x & 63;
    const int ty = threadIdx.x >> 6;      // 0..3 -> channel phase

    // per-thread weights for pixel (n,h,w); read exactly once -> non-temporal
    float wt[36];
    const float* wp = wts + (size_t)((n << 12) + (h << 6) + w) * 36;
#pragma unroll
    for (int j = 0; j < 36; ++j) wt[j] = __builtin_nontemporal_load(wp + j);

    // clamped in-bounds addresses + validity masks for the 3x3 window
    int hcl[3]; bool rok[3];
    int wcl[3]; bool cok[3];
#pragma unroll
    for (int d = 0; d < 3; ++d) {
        int hh = h + d - 1;
        hcl[d] = hh < 0 ? 0 : (hh > 63 ? 63 : hh);
        rok[d] = (hh == hcl[d]);
        int ww = w + d - 1;
        wcl[d] = ww < 0 ? 0 : (ww > 63 ? 63 : ww);
        cok[d] = (ww == wcl[d]);
    }

    const size_t obase = ((size_t)n * C_IN) * (size_t)(128 * 128)
                       + (size_t)(2 * h) * 128 + (size_t)(2 * w);

#pragma unroll 2
    for (int i = 0; i < 64; ++i) {
        const int c = (i << 2) | ty;
        const float* xc = x + ((size_t)n * C_IN + c) * HWD;

        float p[9];
#pragma unroll
        for (int di = 0; di < 3; ++di) {
#pragma unroll
            for (int dj = 0; dj < 3; ++dj) {
                float v = xc[(size_t)hcl[di] * WD + wcl[dj]];   // always in-bounds
                p[di * 3 + dj] = (rok[di] && cok[dj]) ? v : 0.f;
            }
        }

        float a0 = 0.f, a1 = 0.f, a2 = 0.f, a3 = 0.f;
#pragma unroll
        for (int k = 0; k < 9; ++k) {
            a0 = fmaf(p[k], wt[k * 4 + 0], a0);
            a1 = fmaf(p[k], wt[k * 4 + 1], a1);
            a2 = fmaf(p[k], wt[k * 4 + 2], a2);
            a3 = fmaf(p[k], wt[k * 4 + 3], a3);
        }

        float* oc = out + obase + (size_t)c * (128 * 128);
        v2f r0; r0.x = a0; r0.y = a1;   // row 2h,   cols 2w..2w+1
        v2f r1; r1.x = a2; r1.y = a3;   // row 2h+1, cols 2w..2w+1
        __builtin_nontemporal_store(r0, (v2f*)(oc));
        __builtin_nontemporal_store(r1, (v2f*)(oc + 128));
    }
}

// ---------------------------------------------------------------------------
extern "C" void kernel_launch(void* const* d_in, const int* in_sizes, int n_in,
                              void* d_out, int out_size, void* d_ws, size_t ws_size,
                              hipStream_t stream) {
    (void)in_sizes; (void)n_in; (void)out_size; (void)ws_size;
    const float* x   = (const float*)d_in[0];
    const float* w0  = (const float*)d_in[1];
    const float* b0  = (const float*)d_in[2];
    const float* g0  = (const float*)d_in[3];
    const float* bb0 = (const float*)d_in[4];
    const float* m0  = (const float*)d_in[5];
    const float* v0  = (const float*)d_in[6];
    const float* w1  = (const float*)d_in[7];
    const float* b1  = (const float*)d_in[8];
    const float* g1  = (const float*)d_in[9];
    const float* bb1 = (const float*)d_in[10];
    const float* m1  = (const float*)d_in[11];
    const float* v1  = (const float*)d_in[12];

    char* ws = (char*)d_ws;
    float*    gamma0 = (float*)(ws + 0);
    float*    beta0  = (float*)(ws + 256);
    float*    gamma1 = (float*)(ws + 512);
    float*    beta1  = (float*)(ws + 768);
    _Float16* w0h    = (_Float16*)(ws + 1024);
    _Float16* w1h    = (_Float16*)(ws + 33792);
    float*    wts    = (float*)(ws + 40960);   // 32768 * 36 f32

    carafe_prep<<<1, 256, 0, stream>>>(w0, b0, g0, bb0, m0, v0,
                                       w1, b1, g1, bb1, m1, v1,
                                       gamma0, beta0, gamma1, beta1, w0h, w1h);
    carafe_enc<<<256, 256, 0, stream>>>(x, w0h, w1h, gamma0, beta0, gamma1, beta1, wts);
    carafe_apply<<<512, 256, 0, stream>>>(x, wts, (float*)d_out);
}
